// LoRALinear_13838384628026
// MI455X (gfx1250) — compile-verified, matrix operand order checked by reading, not run on hardware
//
#include <hip/hip_runtime.h>

// ---------------- problem constants ----------------
constexpr int HD    = 4096;           // hidden dim (H)
constexpr int RANK  = 16;             // lora rank
constexpr float LORA_SCALE = 1.0f;    // ALPHA / R = 16/16
constexpr int MTOT  = 4 * 2048;       // B * S = 8192 rows
// ---------------- tiling ----------------
constexpr int TBM = 128;              // block tile M
constexpr int TBN = 128;              // block tile N
constexpr int TBK = 64;               // block tile K
constexpr int LDK = TBK + 8;          // padded LDS K-stride (bf16), 144B rows (16B aligned)

typedef __attribute__((ext_vector_type(4)))  __bf16 bf16x4;
typedef __attribute__((ext_vector_type(8)))  __bf16 bf16x8;
typedef __attribute__((ext_vector_type(16))) __bf16 bf16x16;
typedef __attribute__((ext_vector_type(8)))  float  v8f;
typedef int v4i __attribute__((vector_size(16)));   // matches builtin param type

#if __has_builtin(__builtin_amdgcn_global_load_async_to_lds_b128)
#define HAVE_ASYNC_LDS 1
#endif

// 16-byte global -> LDS copy: async DMA (ASYNCcnt) when available, else sync.
__device__ __forceinline__ void copy16_g2l(const __bf16* g, __bf16* l) {
#ifdef HAVE_ASYNC_LDS
  __builtin_amdgcn_global_load_async_to_lds_b128(
      (__attribute__((address_space(1))) v4i*)g,
      (__attribute__((address_space(3))) v4i*)l, 0, 0);
#else
  *(bf16x8*)l = *(const bf16x8*)g;
#endif
}

__device__ __forceinline__ void wait_async_lds() {
#ifdef HAVE_ASYNC_LDS
#if __has_builtin(__builtin_amdgcn_s_wait_asynccnt)
  __builtin_amdgcn_s_wait_asynccnt(0);
#else
  asm volatile("s_wait_asynccnt 0" ::: "memory");
#endif
#endif
}

__constant__ float NF4_LUT[16] = {
    -1.0f, -0.6961928009986877f, -0.5250730514526367f, -0.39491748809814453f,
    -0.28444138169288635f, -0.18477343022823334f, -0.09105003625154495f, 0.0f,
    0.07958029955625534f, 0.16093020141124725f, 0.24611230194568634f,
    0.33791524171829224f, 0.44070982933044434f, 0.5626170039176941f,
    0.7229568362236023f, 1.0f };

// ---------------------------------------------------------------------------
// Kernel 1: W_eff[n][k] = NF4[q]*scale[n][k/64] + (ALPHA/R)*sum_r B[n][r]*A[r][k]
// emitted as bf16 row-major [HD][HD] into workspace.
// ---------------------------------------------------------------------------
__global__ __launch_bounds__(256) void nf4_lora_fuse_kernel(
    const int*   __restrict__ q,       // [HD][HD/2], byte in low 8 bits
    const float* __restrict__ scales,  // [HD][HD/64]
    const float* __restrict__ la,      // [RANK][HD]
    const float* __restrict__ lb,      // [HD][RANK]
    __bf16*      __restrict__ weff)    // [HD][HD]
{
  const int n  = blockIdx.x;
  const int t  = threadIdx.x;
  const int k0 = t << 4;                        // 16 k per thread

  float bv[RANK];
  #pragma unroll
  for (int r = 0; r < RANK; r += 4) {
    float4 b4 = *(const float4*)(lb + n * RANK + r);
    bv[r] = b4.x; bv[r + 1] = b4.y; bv[r + 2] = b4.z; bv[r + 3] = b4.w;
  }
  const float scale = scales[n * (HD / 64) + (k0 >> 6)];

  int4 q0 = *(const int4*)(q + (size_t)n * (HD / 2) + (k0 >> 1));
  int4 q1 = *(const int4*)(q + (size_t)n * (HD / 2) + (k0 >> 1) + 4);
  int qi[8] = {q0.x, q0.y, q0.z, q0.w, q1.x, q1.y, q1.z, q1.w};

  float w[16];
  #pragma unroll
  for (int i = 0; i < 8; ++i) {
    w[2 * i]     = NF4_LUT[qi[i] & 0xF] * scale;         // low nibble -> element 2k
    w[2 * i + 1] = NF4_LUT[(qi[i] >> 4) & 0xF] * scale;  // high nibble -> 2k+1
  }

  #pragma unroll
  for (int r = 0; r < RANK; ++r) {
    const float4* ar = (const float4*)(la + (size_t)r * HD + k0);
    const float br = LORA_SCALE * bv[r];
    #pragma unroll
    for (int j4 = 0; j4 < 4; ++j4) {
      float4 a4 = ar[j4];
      w[4 * j4 + 0] += br * a4.x;
      w[4 * j4 + 1] += br * a4.y;
      w[4 * j4 + 2] += br * a4.z;
      w[4 * j4 + 3] += br * a4.w;
    }
  }

  bf16x8 lo, hi;
  #pragma unroll
  for (int j = 0; j < 8; ++j) { lo[j] = (__bf16)w[j]; hi[j] = (__bf16)w[j + 8]; }
  *(bf16x8*)(weff + (size_t)n * HD + k0)     = lo;
  *(bf16x8*)(weff + (size_t)n * HD + k0 + 8) = hi;
}

// ---------------------------------------------------------------------------
// LDS fragment load, 16-bit A/B 16x32 VGPR layout:
// lane%16 = row, lanes 0-15 hold K {0..7,16..23}, lanes 16-31 hold K {8..15,24..31}.
// ---------------------------------------------------------------------------
__device__ __forceinline__ bf16x16 load_frag(const __bf16* p) {
  bf16x8 lo = *(const bf16x8*)p;
  bf16x8 hi = *(const bf16x8*)(p + 16);
  return __builtin_shufflevector(lo, hi, 0, 1, 2, 3, 4, 5, 6, 7,
                                         8, 9, 10, 11, 12, 13, 14, 15);
}

// ---------------------------------------------------------------------------
// Kernel 2: out = x(->bf16) @ W_eff^T, fp32 accumulate.
// 128x128 block tile, 8 waves (2Mx4N), wave tile 64x32 = 4x2 WMMA tiles.
// LDS ping-pong double buffer; W staged by async-to-LDS DMA; x staged through
// registers (global loads issued one tile ahead, cvt+ds_store after compute).
// ---------------------------------------------------------------------------
__global__ __launch_bounds__(256) void nf4_lora_gemm_kernel(
    const float* __restrict__ x,       // [MTOT][HD] fp32
    const __bf16* __restrict__ w,      // [HD][HD] bf16 (W_eff, L2-resident)
    float* __restrict__ out)           // [MTOT][HD] fp32
{
  __shared__ __bf16 xs[2][TBM * LDK];
  __shared__ __bf16 wsh[2][TBN * LDK];

  const int m0   = blockIdx.y * TBM;
  const int n0   = blockIdx.x * TBN;
  const int t    = threadIdx.x;
  const int lane = t & 31;
  const int wave = t >> 5;
  const int wm   = wave & 1;                // M half: 64 rows
  const int wn   = wave >> 1;               // N quarter: 32 cols
  const int lrow = lane & 15;               // row within 16x16 tile
  const int ksel = (lane >> 4) << 3;        // K octet select: 0 or 8
  const int msel = (lane >> 4) << 3;        // C/D: upper lanes hold M+8

  // per-thread staging coordinates
  const int xrow = t >> 4;                  // x: rows 0..15 (+16 per step)
  const int xc4  = (t & 15) << 2;           // x: float4 column
  const int wrow = t >> 3;                  // w: rows 0..31 (+32 per step)
  const int wcc  = (t & 7) << 3;            // w: bf16x8 column

  v8f acc[4][2];
  #pragma unroll
  for (int i = 0; i < 4; ++i)
    #pragma unroll
    for (int j = 0; j < 2; ++j)
      acc[i][j] = (v8f){0.f, 0.f, 0.f, 0.f, 0.f, 0.f, 0.f, 0.f};

  float4 xr[8];

  // ---- prologue: fill buffer 0 ----
  #pragma unroll
  for (int s = 0; s < 4; ++s)
    copy16_g2l(w + (size_t)(n0 + wrow + s * 32) * HD + wcc,
               &wsh[0][(wrow + s * 32) * LDK + wcc]);
  #pragma unroll
  for (int s = 0; s < 8; ++s)
    xr[s] = *(const float4*)(x + (size_t)(m0 + xrow + s * 16) * HD + xc4);
  #pragma unroll
  for (int s = 0; s < 8; ++s) {
    bf16x4 h;
    h[0] = (__bf16)xr[s].x; h[1] = (__bf16)xr[s].y;
    h[2] = (__bf16)xr[s].z; h[3] = (__bf16)xr[s].w;
    *(bf16x4*)(&xs[0][(xrow + s * 16) * LDK + xc4]) = h;
  }
  wait_async_lds();
  __syncthreads();

  int buf = 0;
  for (int k0 = 0; k0 < HD; k0 += TBK) {
    const int nxt = buf ^ 1;
    const bool more = (k0 + TBK) < HD;

    // ---- issue next tile's global traffic before compute ----
    if (more) {
      const int kn = k0 + TBK;
      #pragma unroll
      for (int s = 0; s < 4; ++s)   // async DMA W -> LDS (overlaps WMMA below)
        copy16_g2l(w + (size_t)(n0 + wrow + s * 32) * HD + kn + wcc,
                   &wsh[nxt][(wrow + s * 32) * LDK + wcc]);
      #pragma unroll
      for (int s = 0; s < 8; ++s)   // x -> registers (latency hidden by WMMA)
        xr[s] = *(const float4*)(x + (size_t)(m0 + xrow + s * 16) * HD + kn + xc4);
      if (kn + TBK < HD)            // L2 prefetch one more tile ahead
        __builtin_prefetch((const char*)(x + (size_t)(m0 + (t >> 1)) * HD +
                                         kn + TBK + ((t & 1) << 5)), 0, 1);
    }

    // ---- compute from current buffer: 16 WMMAs per wave ----
    const __bf16* xb = xs[buf];
    const __bf16* wb = wsh[buf];
    #pragma unroll
    for (int kk = 0; kk < TBK; kk += 32) {
      bf16x16 afrag[4], bfrag[2];
      #pragma unroll
      for (int ti = 0; ti < 4; ++ti)
        afrag[ti] = load_frag(xb + (wm * 64 + ti * 16 + lrow) * LDK + kk + ksel);
      #pragma unroll
      for (int tj = 0; tj < 2; ++tj)
        bfrag[tj] = load_frag(wb + (wn * 32 + tj * 16 + lrow) * LDK + kk + ksel);
      #pragma unroll
      for (int ti = 0; ti < 4; ++ti)
        #pragma unroll
        for (int tj = 0; tj < 2; ++tj)
          acc[ti][tj] = __builtin_amdgcn_wmma_f32_16x16x32_bf16(
              false, afrag[ti], false, bfrag[tj],
              (short)0, acc[ti][tj], false, false);
    }

    // ---- finish staging next buffer, then sync ----
    if (more) {
      #pragma unroll
      for (int s = 0; s < 8; ++s) {
        bf16x4 h;
        h[0] = (__bf16)xr[s].x; h[1] = (__bf16)xr[s].y;
        h[2] = (__bf16)xr[s].z; h[3] = (__bf16)xr[s].w;
        *(bf16x4*)(&xs[nxt][(xrow + s * 16) * LDK + xc4]) = h;
      }
    }
    wait_async_lds();
    __syncthreads();
    buf = nxt;
  }

  // ---- store C: lane%16 = N, VGPR r = M (+8 for upper lane half) ----
  #pragma unroll
  for (int ti = 0; ti < 4; ++ti) {
    #pragma unroll
    for (int tj = 0; tj < 2; ++tj) {
      int mb = m0 + wm * 64 + ti * 16 + msel;
      int nn = n0 + wn * 32 + tj * 16 + lrow;
      float* op = out + (size_t)mb * HD + nn;
      #pragma unroll
      for (int r = 0; r < 8; ++r)
        op[(size_t)r * HD] = acc[ti][tj][r];
    }
  }
}

// ---------------------------------------------------------------------------
extern "C" void kernel_launch(void* const* d_in, const int* in_sizes, int n_in,
                              void* d_out, int out_size, void* d_ws, size_t ws_size,
                              hipStream_t stream) {
  const float* x      = (const float*)d_in[0];   // [4,2048,4096] fp32
  const int*   qpk    = (const int*)d_in[1];     // [4096,2048] int32
  const float* scales = (const float*)d_in[2];   // [4096,64] fp32
  const float* la     = (const float*)d_in[3];   // [16,4096] fp32
  const float* lb     = (const float*)d_in[4];   // [4096,16] fp32
  float* out = (float*)d_out;                    // [4,2048,4096] fp32
  __bf16* weff = (__bf16*)d_ws;                  // 32 MB bf16 W_eff scratch

  // Pass 1: dequant NF4 + fold rank-16 LoRA into bf16 effective weights.
  nf4_lora_fuse_kernel<<<dim3(HD), dim3(256), 0, stream>>>(qpk, scales, la, lb, weff);

  // Pass 2: compute-bound bf16 WMMA GEMM, fp32 accumulate, async-pipelined.
  dim3 grid(HD / TBN, MTOT / TBM);               // 32 x 64 workgroups
  nf4_lora_gemm_kernel<<<grid, dim3(256), 0, stream>>>(x, weff, out);
}
